// CliffordAttention_86715389706730
// MI455X (gfx1250) — matrix operand order repaired and checked by reading, not compiled, not run on hardware
//
#include <hip/hip_runtime.h>
#include <hip/hip_bf16.h>

typedef __attribute__((ext_vector_type(16))) _Float16 v16h;
typedef __attribute__((ext_vector_type(8)))  float    v8f;

#define Bc 2
#define Lc 4096
#define Hc 8
#define Fc 64              // CD*NB per head
#define ROWS (Bc*Lc*Hc)    // 65536 projection rows

// ---------------------------------------------------------------------------
// Kernel 1: QKV projection. q/k/v = xh @ W^T + b, signs*scale folded into q.
// One wave handles 16 rows x 64 outputs x 3 matrices via v_wmma_f32_16x16x32_f16.
// Output layout: [b*H+h][l][64] as f16 (attention-friendly).
// ---------------------------------------------------------------------------
__global__ __launch_bounds__(128) void clifford_proj_qkv(
    const float* __restrict__ x,
    const float* __restrict__ Wq, const float* __restrict__ bq,
    const float* __restrict__ Wk, const float* __restrict__ bk,
    const float* __restrict__ Wv, const float* __restrict__ bv,
    _Float16* __restrict__ qh, _Float16* __restrict__ kh, _Float16* __restrict__ vh)
{
    __shared__ _Float16 sW[3][64 * 64];
    __shared__ float    sb[3][64];

    const int tid = threadIdx.x;
    const int lane = tid & 31, wv = tid >> 5;
    const int hh = lane >> 4, lm = lane & 15;

    for (int i = tid; i < 64 * 64; i += 128) {
        sW[0][i] = (_Float16)Wq[i];
        sW[1][i] = (_Float16)Wk[i];
        sW[2][i] = (_Float16)Wv[i];
    }
    for (int i = tid; i < 64; i += 128) {
        sb[0][i] = bq[i]; sb[1][i] = bk[i]; sb[2][i] = bv[i];
    }
    __syncthreads();

    const long r0 = ((long)blockIdx.x * 4 + wv) * 16;   // first of 16 rows
    const float* xA = x + (r0 + lm) * 64;

    // A-operands (16x32 f16 tiles): kstep0 = feats 0..31, kstep1 = 32..63
    v16h a0, a1;
    #pragma unroll
    for (int j = 0; j < 8; ++j) {
        int k = (hh << 4) + (j << 1);
        a0[2*j]   = (_Float16)xA[k];
        a0[2*j+1] = (_Float16)xA[k + 1];
        a1[2*j]   = (_Float16)xA[32 + k];
        a1[2*j+1] = (_Float16)xA[32 + k + 1];
    }

    _Float16* outs[3] = { qh, kh, vh };

    #pragma unroll
    for (int mm = 0; mm < 3; ++mm) {
        #pragma unroll
        for (int t = 0; t < 4; ++t) {
            const int n = t * 16 + lm;              // output feature
            v16h b0, b1;
            #pragma unroll
            for (int j = 0; j < 8; ++j) {
                int k = (hh << 4) + (j << 1);
                const _Float16* wp = &sW[mm][n * 64 + k];
                b0[2*j]   = wp[0];  b0[2*j+1] = wp[1];
                b1[2*j]   = wp[32]; b1[2*j+1] = wp[33];
            }
            v8f c = {};
            c = __builtin_amdgcn_wmma_f32_16x16x32_f16(false, a0, false, b0, (short)0, c, false, false);
            c = __builtin_amdgcn_wmma_f32_16x16x32_f16(false, a1, false, b1, (short)0, c, false, false);

            const float bias = sb[mm][n];
            // signs = [1,-1,1,1,-1,-1,1,-1] -> negative at i in {1,4,5,7} = mask 0xB2
            float mult = 1.0f;
            if (mm == 0) mult = (((0xB2 >> (n & 7)) & 1) ? -0.125f : 0.125f);

            #pragma unroll
            for (int r = 0; r < 8; ++r) {
                const long rr = r0 + r + 8 * hh;        // flat (b,l,h) row
                const int b_ = (int)(rr >> 15);
                const int l_ = (int)((rr >> 3) & 4095);
                const int h_ = (int)(rr & 7);
                float val = c[r] + bias;
                if (mm == 0) val *= mult;
                const long dst = ((long)(b_ * Hc + h_) * Lc + l_) * 64 + n;
                outs[mm][dst] = (_Float16)val;
            }
        }
    }
}

// ---------------------------------------------------------------------------
// Kernel 2: flash attention, double-buffered with CDNA5 async LDS copies.
// Block = 4 waves x 32; each wave owns 16 query rows; 32 keys per step.
// K tiles staged via global_load_async_to_lds_b128 (ASYNCcnt path); V tiles
// loaded into regs early and stored transposed late (software transpose).
// 8 WMMAs per step: 4 for S = Qw K^T, 4 for O += P V.
// ---------------------------------------------------------------------------
__global__ __launch_bounds__(128) void clifford_flash_attn(
    const _Float16* __restrict__ qh, const _Float16* __restrict__ kh,
    const _Float16* __restrict__ vh, float* __restrict__ out)
{
    __shared__ _Float16 sK[2][32 * 64];    // [buf][key][feat]
    __shared__ _Float16 sVT[2][64 * 32];   // [buf][feat][key]
    __shared__ _Float16 sP[4][16 * 32];    // per-wave P scratch [m][k]

    const int tid = threadIdx.x;
    const int lane = tid & 31, wv = tid >> 5;
    const int hh = lane >> 4, lm = lane & 15;
    const int bh = blockIdx.y;
    const long base = (long)bh * (Lc * 64);
    const int q0 = blockIdx.x * 64 + wv * 16;

    // V-stage decomposition: thread covers key = tid>>2, feats fbase..fbase+15
    const int vkey = tid >> 2, vfb = (tid & 3) * 16;

    // Q A-operands for this wave's 16 rows (signs & scale already folded in)
    const _Float16* qrow = qh + base + (long)(q0 + lm) * 64;
    v16h aq0, aq1;
    #pragma unroll
    for (int j = 0; j < 8; ++j) {
        int k = (hh << 4) + (j << 1);
        aq0[2*j]   = qrow[k];      aq0[2*j+1] = qrow[k + 1];
        aq1[2*j]   = qrow[32 + k]; aq1[2*j+1] = qrow[32 + k + 1];
    }

    v8f oacc[4];
    float mrow[8], lrow[8];
    #pragma unroll
    for (int c4 = 0; c4 < 4; ++c4) oacc[c4] = (v8f){};
    #pragma unroll
    for (int r = 0; r < 8; ++r) { mrow[r] = -1e30f; lrow[r] = 0.0f; }

    union VStage { uint4 u[2]; _Float16 h[16]; };

    // --- prologue: stage tile 0 into buffer 0 ---
    {
        unsigned ldsK = (unsigned)(size_t)(&sK[0][0]) + (unsigned)tid * 16u;
        unsigned long long ga = (unsigned long long)(const void*)(kh + base) +
                                (unsigned long long)tid * 16u;
        asm volatile("global_load_async_to_lds_b128 %0, %1, off"
                     :: "v"(ldsK), "v"(ga) : "memory");
        asm volatile("global_load_async_to_lds_b128 %0, %1, off offset:2048"
                     :: "v"(ldsK), "v"(ga) : "memory");

        VStage vs;
        const uint4* vsrc = (const uint4*)(vh + base) + tid * 2;
        vs.u[0] = vsrc[0]; vs.u[1] = vsrc[1];
        #pragma unroll
        for (int i = 0; i < 16; ++i) sVT[0][(vfb + i) * 32 + vkey] = vs.h[i];

        asm volatile("s_wait_asynccnt 0x0" ::: "memory");
        __syncthreads();
    }

    for (int kt = 0; kt < Lc / 32; ++kt) {
        const int p = kt & 1, pn = p ^ 1;
        int next = kt + 1; if (next > Lc / 32 - 1) next = Lc / 32 - 1;

        // --- (1) issue async K copy + V reg loads for tile `next` into buf pn ---
        {
            unsigned ldsK = (unsigned)(size_t)(&sK[pn][0]) + (unsigned)tid * 16u;
            unsigned long long ga =
                (unsigned long long)(const void*)(kh + base + (long)next * 2048) +
                (unsigned long long)tid * 16u;
            asm volatile("global_load_async_to_lds_b128 %0, %1, off"
                         :: "v"(ldsK), "v"(ga) : "memory");
            asm volatile("global_load_async_to_lds_b128 %0, %1, off offset:2048"
                         :: "v"(ldsK), "v"(ga) : "memory");
        }
        VStage vs;
        {
            const uint4* vsrc = (const uint4*)(vh + base + (long)next * 2048) + tid * 2;
            vs.u[0] = vsrc[0]; vs.u[1] = vsrc[1];
        }

        // --- (2) compute on buffer p ---
        const _Float16* Kc = sK[p];
        const _Float16* Vc = sVT[p];

        v8f s0 = {}, s1 = {};
        {
            v16h b0, b1;
            #pragma unroll
            for (int j = 0; j < 8; ++j) {
                int k = (hh << 4) + (j << 1);
                const _Float16* pk = &Kc[lm * 64 + k];
                b0[2*j] = pk[0];  b0[2*j+1] = pk[1];
                b1[2*j] = pk[32]; b1[2*j+1] = pk[33];
            }
            s0 = __builtin_amdgcn_wmma_f32_16x16x32_f16(false, aq0, false, b0, (short)0, s0, false, false);
            s0 = __builtin_amdgcn_wmma_f32_16x16x32_f16(false, aq1, false, b1, (short)0, s0, false, false);
        }
        {
            v16h b0, b1;
            #pragma unroll
            for (int j = 0; j < 8; ++j) {
                int k = (hh << 4) + (j << 1);
                const _Float16* pk = &Kc[(16 + lm) * 64 + k];
                b0[2*j] = pk[0];  b0[2*j+1] = pk[1];
                b1[2*j] = pk[32]; b1[2*j+1] = pk[33];
            }
            s1 = __builtin_amdgcn_wmma_f32_16x16x32_f16(false, aq0, false, b0, (short)0, s1, false, false);
            s1 = __builtin_amdgcn_wmma_f32_16x16x32_f16(false, aq1, false, b1, (short)0, s1, false, false);
        }

        // --- online softmax: row stats via shfl_xor within 16-lane halves ---
        float cm[8];
        #pragma unroll
        for (int r = 0; r < 8; ++r) cm[r] = fmaxf(s0[r], s1[r]);
        #pragma unroll
        for (int d = 1; d < 16; d <<= 1) {
            #pragma unroll
            for (int r = 0; r < 8; ++r) cm[r] = fmaxf(cm[r], __shfl_xor(cm[r], d, 32));
        }
        float alpha[8];
        #pragma unroll
        for (int r = 0; r < 8; ++r) {
            float mn = fmaxf(mrow[r], cm[r]);
            alpha[r] = __expf(mrow[r] - mn);
            mrow[r] = mn;
        }
        v8f p0, p1;
        float cs[8];
        #pragma unroll
        for (int r = 0; r < 8; ++r) {
            p0[r] = __expf(s0[r] - mrow[r]);
            p1[r] = __expf(s1[r] - mrow[r]);
            cs[r] = p0[r] + p1[r];
        }
        #pragma unroll
        for (int d = 1; d < 16; d <<= 1) {
            #pragma unroll
            for (int r = 0; r < 8; ++r) cs[r] += __shfl_xor(cs[r], d, 32);
        }
        #pragma unroll
        for (int r = 0; r < 8; ++r) {
            lrow[r] = lrow[r] * alpha[r] + cs[r];
            oacc[0][r] *= alpha[r]; oacc[1][r] *= alpha[r];
            oacc[2][r] *= alpha[r]; oacc[3][r] *= alpha[r];
        }

        // --- C-layout P -> A-layout via per-wave LDS scratch ---
        _Float16* pw = sP[wv];
        #pragma unroll
        for (int r = 0; r < 8; ++r) {
            int m = r + 8 * hh;
            pw[m * 32 + lm]      = (_Float16)p0[r];
            pw[m * 32 + 16 + lm] = (_Float16)p1[r];
        }
        asm volatile("" ::: "memory");   // keep DS stores before DS loads
        v16h pa;
        #pragma unroll
        for (int j = 0; j < 8; ++j) {
            int k = (hh << 4) + (j << 1);
            const _Float16* pp = &pw[lm * 32 + k];
            pa[2*j] = pp[0]; pa[2*j+1] = pp[1];
        }

        // --- O += P @ V over 4 feature chunks ---
        #pragma unroll
        for (int c4 = 0; c4 < 4; ++c4) {
            v16h bvv;
            #pragma unroll
            for (int j = 0; j < 8; ++j) {
                int k = (hh << 4) + (j << 1);        // key index pair
                const _Float16* vp = &Vc[(c4 * 16 + lm) * 32 + k];
                bvv[2*j] = vp[0]; bvv[2*j+1] = vp[1];
            }
            oacc[c4] = __builtin_amdgcn_wmma_f32_16x16x32_f16(false, pa, false, bvv, (short)0, oacc[c4], false, false);
        }

        // --- (3) store next V tile transposed into buf pn ---
        #pragma unroll
        for (int i = 0; i < 16; ++i) sVT[pn][(vfb + i) * 32 + vkey] = vs.h[i];

        // --- (4) my async K copy done + all waves finished this step ---
        asm volatile("s_wait_asynccnt 0x0" ::: "memory");
        __syncthreads();
    }

    // --- epilogue: normalize and store (coalesced in n) ---
    const int b_ = bh >> 3, h_ = bh & 7;
    #pragma unroll
    for (int r = 0; r < 8; ++r) {
        const float inv = 1.0f / lrow[r];
        const int l_ = q0 + r + 8 * hh;
        float* orow = out + ((long)(b_ * Lc + l_) * (Hc * 64)) + h_ * 64;
        #pragma unroll
        for (int c4 = 0; c4 < 4; ++c4) orow[c4 * 16 + lm] = oacc[c4][r] * inv;
    }
}

// ---------------------------------------------------------------------------
extern "C" void kernel_launch(void* const* d_in, const int* in_sizes, int n_in,
                              void* d_out, int out_size, void* d_ws, size_t ws_size,
                              hipStream_t stream) {
    const float* x  = (const float*)d_in[0];
    const float* Wq = (const float*)d_in[1];
    const float* bq = (const float*)d_in[2];
    const float* Wk = (const float*)d_in[3];
    const float* bk = (const float*)d_in[4];
    const float* Wv = (const float*)d_in[5];
    const float* bv = (const float*)d_in[6];
    float* out = (float*)d_out;

    const size_t perBuf = (size_t)Bc * Hc * Lc * 64;   // 4,194,304 halves = 8 MB
    _Float16* qh = (_Float16*)d_ws;
    _Float16* kh = qh + perBuf;
    _Float16* vh = kh + perBuf;

    // Projection: 65536 rows / 16 rows per wave / 4 waves per block = 1024 blocks
    clifford_proj_qkv<<<dim3(ROWS / 64), dim3(128), 0, stream>>>(
        x, Wq, bq, Wk, bk, Wv, bv, qh, kh, vh);

    // Attention: grid.x = query tiles of 64 rows, grid.y = B*H
    clifford_flash_attn<<<dim3(Lc / 64, Bc * Hc), dim3(128), 0, stream>>>(
        qh, kh, vh, out);
}
